// MultiHeadAttentionBlock_4294967296499
// MI455X (gfx1250) — matrix operand order repaired and checked by reading, not compiled, not run on hardware
//
#include <hip/hip_runtime.h>
#include <math.h>

#define D_MODEL 1024
#define N_HEADS 16
#define DK      64
#define SEQ     2048
#define NBATCH  2
#define NROWS   (NBATCH * SEQ)      // 4096
#define ROPE_BASE 500000.0f
#define RMS_EPS   1e-10f

typedef __attribute__((ext_vector_type(16))) __bf16 v16bf;
typedef __attribute__((ext_vector_type(8)))  float  v8f;

static __device__ __forceinline__ v8f wmma_bf16(v16bf a, v16bf b, v8f c) {
  // D = A(16x32 bf16) * B(32x16 bf16) + C(16x16 f32)
  return __builtin_amdgcn_wmma_f32_16x16x32_bf16(false, a, false, b,
                                                 (short)0, c, false, false);
}

union BfPack { unsigned u; __bf16 h[2]; };

// ---- fragment loaders (CDNA5 wave32 WMMA layouts) -------------------------

// A 16x32 bf16 from bf16 memory, element (m,k) at base[m*lda + k].
static __device__ __forceinline__ v16bf load_a_bf16(const __bf16* base, int lda) {
  int lane = threadIdx.x & 31;
  int m = lane & 15, g = lane >> 4;
  const __bf16* p = base + m * lda + 16 * g;
  v16bf a;
#pragma unroll
  for (int v = 0; v < 8; ++v) {
    int kb = 2 * v;   // V0..7: K=2v(+16g)
    BfPack c; c.u = *(const unsigned*)(p + kb);
    a[2 * v] = c.h[0]; a[2 * v + 1] = c.h[1];
  }
  return a;
}

// A 16x32 from fp32 memory with inline bf16 conversion.
static __device__ __forceinline__ v16bf load_a_f32cvt(const float* base, int lda) {
  int lane = threadIdx.x & 31;
  int m = lane & 15, g = lane >> 4;
  const float* p = base + m * lda + 16 * g;
  v16bf a;
#pragma unroll
  for (int v = 0; v < 8; ++v) {
    int kb = 2 * v;
    float2 f = *(const float2*)(p + kb);
    a[2 * v] = (__bf16)f.x; a[2 * v + 1] = (__bf16)f.y;
  }
  return a;
}

// B 32x16 bf16, element (k,n) at base[n*ld + k]  (k contiguous).
static __device__ __forceinline__ v16bf load_b_kcontig(const __bf16* base, int ld) {
  int lane = threadIdx.x & 31;
  int n = lane & 15, g = lane >> 4;
  const __bf16* p = base + n * ld + 16 * g;
  v16bf b;
#pragma unroll
  for (int v = 0; v < 8; ++v) {
    BfPack c; c.u = *(const unsigned*)(p + 2 * v);
    b[2 * v] = c.h[0]; b[2 * v + 1] = c.h[1];
  }
  return b;
}

// B 32x16 bf16, element (k,n) at base[k*ld + n]  (n contiguous).
static __device__ __forceinline__ v16bf load_b_ncontig(const __bf16* base, int ld) {
  int lane = threadIdx.x & 31;
  int n = lane & 15, g = lane >> 4;
  const __bf16* p = base + n + 16 * g * ld;
  v16bf b;
#pragma unroll
  for (int v = 0; v < 8; ++v) {
    b[2 * v]     = p[(2 * v) * ld];
    b[2 * v + 1] = p[(2 * v + 1) * ld];
  }
  return b;
}

// ---- kernel 1: fp32 -> bf16 weight conversion -----------------------------

__global__ __launch_bounds__(256) void cvt_f32_bf16(const float* __restrict__ in,
                                                    __bf16* __restrict__ out, int n4) {
  int i = blockIdx.x * blockDim.x + threadIdx.x;
  if (i < n4) {
    float4 f = ((const float4*)in)[i];
    BfPack p0, p1;
    p0.h[0] = (__bf16)f.x; p0.h[1] = (__bf16)f.y;
    p1.h[0] = (__bf16)f.z; p1.h[1] = (__bf16)f.w;
    ((uint2*)out)[i] = make_uint2(p0.u, p1.u);
  }
}

// ---- kernel 2: fused QKV projection + RMS-norm + RoPE ---------------------
// grid: (heads=16, NROWS/128=32, mode=3); block 256 (8 waves, 16 rows each).
// out layout: [B, H, S, DK] bf16.

__global__ __launch_bounds__(256) void qkv_proj_kernel(
    const float* __restrict__ xq, const float* __restrict__ xk,
    const float* __restrict__ xv,
    const __bf16* __restrict__ wq, const __bf16* __restrict__ wk,
    const __bf16* __restrict__ wv,
    const float* __restrict__ qnw, const float* __restrict__ knw,
    __bf16* __restrict__ qout, __bf16* __restrict__ kout,
    __bf16* __restrict__ vout) {
  int mode = blockIdx.z;
  const float*  x  = (mode == 0) ? xq : (mode == 1) ? xk : xv;
  const __bf16* w  = (mode == 0) ? wq : (mode == 1) ? wk : wv;
  __bf16* outp     = (mode == 0) ? qout : (mode == 1) ? kout : vout;
  const float* nw  = (mode == 0) ? qnw : knw;

  int h    = blockIdx.x;
  int wave = threadIdx.x >> 5;
  int lane = threadIdx.x & 31;
  int g = lane >> 4, nl = lane & 15;
  int row0 = blockIdx.y * 128 + wave * 16;
  int n0 = h * DK;

  v8f zero = {0, 0, 0, 0, 0, 0, 0, 0};
  v8f acc[4] = {zero, zero, zero, zero};

  for (int kk = 0; kk < D_MODEL; kk += 32) {
    v16bf a = load_a_f32cvt(x + (size_t)row0 * D_MODEL + kk, D_MODEL);
#pragma unroll
    for (int t = 0; t < 4; ++t) {
      v16bf b = load_b_kcontig(w + (size_t)(n0 + t * 16) * D_MODEL + kk, D_MODEL);
      acc[t] = wmma_bf16(a, b, acc[t]);
    }
  }

  if (mode < 2) {
    // RMS norm over DK=64 per row (rows live in VGPR index j within lane group g)
    float rs[8];
#pragma unroll
    for (int j = 0; j < 8; ++j) {
      float s = 0.f;
#pragma unroll
      for (int t = 0; t < 4; ++t) { float v = acc[t][j]; s += v * v; }
      s += __shfl_xor(s, 1, 32); s += __shfl_xor(s, 2, 32);
      s += __shfl_xor(s, 4, 32); s += __shfl_xor(s, 8, 32);
      rs[j] = rsqrtf(s * (1.0f / DK) + RMS_EPS);
    }
    float nw0 = nw[nl], nw1 = nw[nl + 16], nw2 = nw[nl + 32], nw3 = nw[nl + 48];
#pragma unroll
    for (int j = 0; j < 8; ++j) {
      acc[0][j] *= rs[j] * nw0; acc[1][j] *= rs[j] * nw1;
      acc[2][j] *= rs[j] * nw2; acc[3][j] *= rs[j] * nw3;
    }
    // RoPE: inv_freq[i] = base^(-i/32), i = col % 32; cols per lane: nl, nl+16 (+32 same angle)
    float if0 = __powf(ROPE_BASE, -(float)nl / 32.0f);
    float if1 = __powf(ROPE_BASE, -(float)(nl + 16) / 32.0f);
#pragma unroll
    for (int j = 0; j < 8; ++j) {
      int row = row0 + j + 8 * g;
      float spos = (float)(row & (SEQ - 1));
      float a0 = spos * if0, a1 = spos * if1;
      float c0 = __cosf(a0), s0 = __sinf(a0);
      float c1 = __cosf(a1), s1 = __sinf(a1);
      float v0 = acc[0][j], v1 = acc[1][j], v2 = acc[2][j], v3 = acc[3][j];
      acc[0][j] = v0 * c0 - v2 * s0;   // n < 32 : rot = -x[n+32]
      acc[1][j] = v1 * c1 - v3 * s1;
      acc[2][j] = v2 * c0 + v0 * s0;   // n >= 32: rot = +x[n-32]
      acc[3][j] = v3 * c1 + v1 * s1;
    }
  }

#pragma unroll
  for (int j = 0; j < 8; ++j) {
    int row = row0 + j + 8 * g;
    int b = row >> 11, s = row & (SEQ - 1);
    size_t obase = ((size_t)(b * N_HEADS + h) * SEQ + s) * DK;
#pragma unroll
    for (int t = 0; t < 4; ++t)
      outp[obase + nl + 16 * t] = (__bf16)acc[t][j];
  }
}

// ---- kernel 3: causal flash attention -------------------------------------
// grid: (SEQ/64=32 query blocks, B*H=32); block 128 (4 waves x 16 query rows).

__global__ __launch_bounds__(128) void flash_attn_kernel(
    const __bf16* __restrict__ qb, const __bf16* __restrict__ kb,
    const __bf16* __restrict__ vb, __bf16* __restrict__ ob) {
  __shared__ __bf16 kt[64 * 64];
  __shared__ __bf16 vt[64 * 64];
  __shared__ __bf16 pt[4][16 * 64];

  int qblk = blockIdx.x;
  int bh   = blockIdx.y;                      // b*N_HEADS + h
  int tid  = threadIdx.x;
  int wave = tid >> 5, lane = tid & 31;
  int g = lane >> 4, nl = lane & 15;
  int bidx = bh >> 4, hidx = bh & 15;

  const __bf16* qbase = qb + (size_t)bh * SEQ * DK;
  const __bf16* kbase = kb + (size_t)bh * SEQ * DK;
  const __bf16* vbase = vb + (size_t)bh * SEQ * DK;

  int qrow0 = qblk * 64 + wave * 16;
  v16bf qf0 = load_a_bf16(qbase + (size_t)qrow0 * DK + 0, DK);
  v16bf qf1 = load_a_bf16(qbase + (size_t)qrow0 * DK + 32, DK);

  v8f zero = {0, 0, 0, 0, 0, 0, 0, 0};
  v8f oacc[4] = {zero, zero, zero, zero};
  float mrow[8], lrow[8];
#pragma unroll
  for (int j = 0; j < 8; ++j) { mrow[j] = -INFINITY; lrow[j] = 0.f; }

  const float scale = 0.125f;                 // 1/sqrt(64)
  int ntiles = qblk + 1;

  for (int tk = 0; tk < ntiles; ++tk) {
    int kv0 = tk * 64;
    __syncthreads();
    {   // stage the first 32x64 (4 KB) of the K and V tiles into LDS
      const uint2* ks = (const uint2*)(kbase + (size_t)kv0 * DK);
      const uint2* vs = (const uint2*)(vbase + (size_t)kv0 * DK);
      uint2* kd = (uint2*)kt; uint2* vd = (uint2*)vt;
#pragma unroll
      for (int i = 0; i < 4; ++i) {
        kd[i * 128 + tid] = ks[i * 128 + tid];
        vd[i * 128 + tid] = vs[i * 128 + tid];
      }
    }
    __syncthreads();

    // scores: Q(16x64) . K^T -> 16x64
    v8f sacc[4] = {zero, zero, zero, zero};
#pragma unroll
    for (int ks = 0; ks < 2; ++ks) {
      v16bf a = ks ? qf1 : qf0;
#pragma unroll
      for (int t = 0; t < 4; ++t) {
        v16bf b = load_b_kcontig(kt + (t * 16) * 64 + ks * 32, 64);
        sacc[t] = wmma_bf16(a, b, sacc[t]);
      }
    }

    // online softmax (per-row state in registers; rows = j + 8g)
    float alpha[8];
#pragma unroll
    for (int j = 0; j < 8; ++j) {
      int row = qrow0 + j + 8 * g;
      float mj = -INFINITY;
#pragma unroll
      for (int t = 0; t < 4; ++t) {
        int col = kv0 + t * 16 + nl;
        float v = sacc[t][j] * scale;
        if (col > row) v = -INFINITY;         // causal mask
        sacc[t][j] = v;
        mj = fmaxf(mj, v);
      }
      mj = fmaxf(mj, __shfl_xor(mj, 1, 32)); mj = fmaxf(mj, __shfl_xor(mj, 2, 32));
      mj = fmaxf(mj, __shfl_xor(mj, 4, 32)); mj = fmaxf(mj, __shfl_xor(mj, 8, 32));
      float mn = fmaxf(mrow[j], mj);
      float al = __expf(mrow[j] - mn);
      float rs = 0.f;
#pragma unroll
      for (int t = 0; t < 4; ++t) {
        float p = __expf(sacc[t][j] - mn);
        pt[wave][(j + 8 * g) * 64 + t * 16 + nl] = (__bf16)p;   // C-layout -> LDS
        rs += p;
      }
      rs += __shfl_xor(rs, 1, 32); rs += __shfl_xor(rs, 2, 32);
      rs += __shfl_xor(rs, 4, 32); rs += __shfl_xor(rs, 8, 32);
      mrow[j] = mn;
      lrow[j] = lrow[j] * al + rs;
      alpha[j] = al;
    }
#pragma unroll
    for (int t = 0; t < 4; ++t)
#pragma unroll
      for (int j = 0; j < 8; ++j) oacc[t][j] *= alpha[j];

    // O += P(16x64) . V(64x64); P re-read from LDS in A-layout (same wave)
#pragma unroll
    for (int ks = 0; ks < 2; ++ks) {
      v16bf a = load_a_bf16(&pt[wave][ks * 32], 64);
#pragma unroll
      for (int t = 0; t < 4; ++t) {
        v16bf b = load_b_ncontig(vt + (ks * 32) * 64 + t * 16, 64);
        oacc[t] = wmma_bf16(a, b, oacc[t]);
      }
    }
  }

  // normalize and store to [B, S, H*DK] bf16 for the output projection
#pragma unroll
  for (int j = 0; j < 8; ++j) {
    float inv = 1.0f / lrow[j];
    int s = qrow0 + j + 8 * g;
    size_t obase = ((size_t)bidx * SEQ + s) * D_MODEL + hidx * DK;
#pragma unroll
    for (int t = 0; t < 4; ++t)
      ob[obase + t * 16 + nl] = (__bf16)(oacc[t][j] * inv);
  }
}

// ---- kernel 4: output projection, fp32 result -----------------------------
// grid: (D_MODEL/64=16, NROWS/128=32); block 256 (8 waves x 16 rows).

__global__ __launch_bounds__(256) void out_proj_kernel(
    const __bf16* __restrict__ xin, const __bf16* __restrict__ wo,
    float* __restrict__ out) {
  int wave = threadIdx.x >> 5, lane = threadIdx.x & 31;
  int g = lane >> 4, nl = lane & 15;
  int row0 = blockIdx.y * 128 + wave * 16;
  int n0 = blockIdx.x * 64;

  v8f zero = {0, 0, 0, 0, 0, 0, 0, 0};
  v8f acc[4] = {zero, zero, zero, zero};

  for (int kk = 0; kk < D_MODEL; kk += 32) {
    v16bf a = load_a_bf16(xin + (size_t)row0 * D_MODEL + kk, D_MODEL);
#pragma unroll
    for (int t = 0; t < 4; ++t) {
      v16bf b = load_b_kcontig(wo + (size_t)(n0 + t * 16) * D_MODEL + kk, D_MODEL);
      acc[t] = wmma_bf16(a, b, acc[t]);
    }
  }
#pragma unroll
  for (int j = 0; j < 8; ++j) {
    int row = row0 + j + 8 * g;
#pragma unroll
    for (int t = 0; t < 4; ++t)
      out[(size_t)row * D_MODEL + n0 + t * 16 + nl] = acc[t][j];
  }
}

// ---- launch ---------------------------------------------------------------

extern "C" void kernel_launch(void* const* d_in, const int* in_sizes, int n_in,
                              void* d_out, int out_size, void* d_ws, size_t ws_size,
                              hipStream_t stream) {
  const float* q   = (const float*)d_in[0];
  const float* k   = (const float*)d_in[1];
  const float* v   = (const float*)d_in[2];
  const float* w_q = (const float*)d_in[3];
  const float* w_k = (const float*)d_in[4];
  const float* w_v = (const float*)d_in[5];
  const float* w_o = (const float*)d_in[6];
  const float* qnw = (const float*)d_in[7];
  const float* knw = (const float*)d_in[8];
  float* out = (float*)d_out;

  char* ws = (char*)d_ws;
  const size_t WELEMS = (size_t)D_MODEL * D_MODEL;      // 1M elems
  const size_t AELEMS = (size_t)NROWS * D_MODEL;        // 4M elems
  __bf16* wq_bf = (__bf16*)(ws + 0 * WELEMS * 2);
  __bf16* wk_bf = (__bf16*)(ws + 1 * WELEMS * 2);
  __bf16* wv_bf = (__bf16*)(ws + 2 * WELEMS * 2);
  __bf16* wo_bf = (__bf16*)(ws + 3 * WELEMS * 2);
  __bf16* q_bf  = (__bf16*)(ws + 4 * WELEMS * 2 + 0 * AELEMS * 2);
  __bf16* k_bf  = (__bf16*)(ws + 4 * WELEMS * 2 + 1 * AELEMS * 2);
  __bf16* v_bf  = (__bf16*)(ws + 4 * WELEMS * 2 + 2 * AELEMS * 2);
  __bf16* a_bf  = (__bf16*)(ws + 4 * WELEMS * 2 + 3 * AELEMS * 2);
  (void)in_sizes; (void)n_in; (void)out_size; (void)ws_size;

  int n4 = (int)(WELEMS / 4);
  dim3 cgrid((n4 + 255) / 256);
  cvt_f32_bf16<<<cgrid, 256, 0, stream>>>(w_q, wq_bf, n4);
  cvt_f32_bf16<<<cgrid, 256, 0, stream>>>(w_k, wk_bf, n4);
  cvt_f32_bf16<<<cgrid, 256, 0, stream>>>(w_v, wv_bf, n4);
  cvt_f32_bf16<<<cgrid, 256, 0, stream>>>(w_o, wo_bf, n4);

  qkv_proj_kernel<<<dim3(N_HEADS, NROWS / 128, 3), 256, 0, stream>>>(
      q, k, v, wq_bf, wk_bf, wv_bf, qnw, knw, q_bf, k_bf, v_bf);

  flash_attn_kernel<<<dim3(SEQ / 64, NBATCH * N_HEADS), 128, 0, stream>>>(
      q_bf, k_bf, v_bf, a_bf);

  out_proj_kernel<<<dim3(D_MODEL / 64, NROWS / 128), 256, 0, stream>>>(
      a_bf, wo_bf, out);
}
